// SovereignLeviathanV2_37125697307214
// MI455X (gfx1250) — matrix operand order repaired and checked by reading, not compiled, not run on hardware
//
#include <hip/hip_runtime.h>
#include <hip/hip_bf16.h>
#include <stdint.h>

// ---------------------------------------------------------------------------
// RNN + top-1 MoE + LM head for MI455X (gfx1250, wave32, WMMA + TDM).
// VOCAB=32000, D=1024, E=4, B=4, S=1024, DF=4096.  Workspace: ~213 MB.
// ---------------------------------------------------------------------------

#define VOCAB 32000
#define DIM   1024
#define NE    4
#define NB    4
#define SEQ   1024
#define DFF   4096

typedef __bf16 bf16;
typedef __attribute__((ext_vector_type(8)))  __bf16 v8bf;
typedef __attribute__((ext_vector_type(16))) __bf16 v16bf;
typedef __attribute__((ext_vector_type(8)))  float  v8f;
typedef __attribute__((ext_vector_type(4)))  unsigned int v4u;
typedef __attribute__((ext_vector_type(8)))  int v8i;
typedef __attribute__((ext_vector_type(4)))  int v4i;

union V16U { v16bf v; v8bf h[2]; };

// ---- WMMA helper ----------------------------------------------------------
__device__ __forceinline__ v8f wmma_bf16(v16bf a, v16bf b, v8f c) {
  return __builtin_amdgcn_wmma_f32_16x16x32_bf16(false, a, false, b,
                                                 (short)0, c, false, false);
}

// A-tile (16x32 bf16, row-major, stride ld): documented per-lane layout.
__device__ __forceinline__ v16bf load_a16(const bf16* base, int ld) {
  int lane = threadIdx.x & 31;
  int m  = lane & 15;
  int ko = (lane & 16) ? 8 : 0;
  const bf16* p = base + m * ld;
  V16U r;
  r.h[0] = *(const v8bf*)(p + ko);
  r.h[1] = *(const v8bf*)(p + 16 + ko);
  return r.v;
}

// B-tile (32x16 bf16, K-major, stride ld): lane%16 = N col, K in table order.
__device__ __forceinline__ v16bf load_b16(const bf16* base, int ld) {
  int lane = threadIdx.x & 31;
  int n  = lane & 15;
  int ko = (lane & 16) ? 8 : 0;
  v16bf r;
  #pragma unroll
  for (int s = 0; s < 16; ++s) {
    int k = (s < 8) ? (ko + s) : (16 + ko + s - 8);
    r[s] = base[(long)k * ld + n];
  }
  return r;
}

// ---- Tensor Data Mover: 2D tile (td1 rows x td0 elems, bf16) -> LDS -------
// Builds D# group0/1 per CDNA5 ISA 8.3/8.4; groups 2/3 zero (2D tile).
__device__ __forceinline__ void tdm_load_2d(unsigned lds, const void* g,
                                            unsigned td0, unsigned td1,
                                            unsigned long stride_elems) {
  unsigned long ga = (unsigned long)g;
  v4u g0;
  g0[0] = 1u;                                              // count=1, user D#
  g0[1] = lds;                                             // LDS byte address
  g0[2] = (unsigned)(ga & 0xFFFFFFFFu);                    // global_addr lo
  g0[3] = (unsigned)((ga >> 32) & 0x01FFFFFFu) | (2u << 30);  // hi | type=2
  v8i g1;
  g1[0] = (int)(1u << 16);                                 // data_size = 2B
  g1[1] = (int)((td0 & 0xFFFFu) << 16);                    // tensor_dim0 lo16
  g1[2] = (int)((td0 >> 16) | ((td1 & 0xFFFFu) << 16));    // d0 hi | d1 lo
  g1[3] = (int)((td1 >> 16) | ((td0 & 0xFFFFu) << 16));    // d1 hi | tile_dim0
  g1[4] = (int)(td1 & 0xFFFFu);                            // tile_dim1 (d2=0)
  g1[5] = (int)(stride_elems & 0xFFFFFFFFu);               // dim0_stride lo
  g1[6] = (int)((stride_elems >> 32) & 0xFFFFu);           // dim0_stride hi
  g1[7] = 0;                                               // dim1_stride = 0
  v4i z4 = {};
#if defined(__clang_major__) && (__clang_major__ >= 23)
  v8i z8 = {};
  __builtin_amdgcn_tensor_load_to_lds(g0, g1, z4, z4, z8, 0);
#else
  __builtin_amdgcn_tensor_load_to_lds(g0, g1, z4, z4, 0);
#endif
}

// ---- small prep kernels ---------------------------------------------------
__global__ void zero_u32_k(unsigned int* p, long n) {
  long i = (long)blockIdx.x * blockDim.x + threadIdx.x;
  if (i < n) p[i] = 0u;
}

__global__ void bias2_k(const float* a, const float* b, float* o, int n) {
  int i = blockIdx.x * blockDim.x + threadIdx.x;
  if (i < n) o[i] = a[i] + b[i];
}

__global__ void cvt_bf_k(const float* __restrict__ src, bf16* __restrict__ dst, long n4) {
  long i = ((long)blockIdx.x * blockDim.x + threadIdx.x) * 4;
  if (i < n4) {
    float4 v = *(const float4*)(src + i);
    dst[i + 0] = (bf16)v.x; dst[i + 1] = (bf16)v.y;
    dst[i + 2] = (bf16)v.z; dst[i + 3] = (bf16)v.w;
  }
}

// dst[c*rows + r] = (bf16) src[r*cols + c]
__global__ __launch_bounds__(256) void transpose_bf_k(
    const float* __restrict__ src, bf16* __restrict__ dst, int rows, int cols) {
  __shared__ bf16 tile[32][33];
  int c0 = blockIdx.x * 32, r0 = blockIdx.y * 32;
  int tx = threadIdx.x & 31, ty = threadIdx.x >> 5;
  #pragma unroll
  for (int i = 0; i < 32; i += 8)
    tile[ty + i][tx] = (bf16)src[(long)(r0 + ty + i) * cols + c0 + tx];
  __syncthreads();
  #pragma unroll
  for (int i = 0; i < 32; i += 8)
    dst[(long)(c0 + ty + i) * rows + r0 + tx] = tile[tx][ty + i];
}

__global__ void embed_k(const int* __restrict__ tokens, const float* __restrict__ emb,
                        bf16* __restrict__ Xbf) {
  int row = blockIdx.x;
  int col = blockIdx.y * blockDim.x + threadIdx.x;
  int tok = tokens[row];
  Xbf[(long)row * DIM + col] = (bf16)emb[(long)tok * DIM + col];
}

// ---- generic bf16 WMMA GEMM with TDM-staged, double-buffered LDS ----------
// C[M,N] = A[M,K] @ B[K,N] (+bias), tile 64x128, BK=32, 8 waves.
// MODE 0: fp32 out.            MODE 1: GELU -> bf16, expert-indexed.
// MODE 2: *gateW[b] -> bf16.   MODE 3: fp32 out, nontemporal (streaming).
#define BM 64
#define BN 128
#define BK 32
#define ABYTES (BM * BK * 2)
#define BBYTES (BK * BN * 2)

template <int MODE>
__global__ __launch_bounds__(256) void gemm_bf16_k(
    const bf16* __restrict__ A, const bf16* __restrict__ Bmat,
    const float* bias, float* __restrict__ Cf, bf16* __restrict__ Cbf,
    int M, int N, int K, int lda, int ldb, int ldc,
    const int* __restrict__ eidx, long strideBe, int biasStrideE,
    long strideAb, long strideCb, const float* __restrict__ gateW) {
  __shared__ bf16 As[2][BM * BK];
  __shared__ bf16 Bs[2][BK * BN];

  int tid = threadIdx.x;
  int bn0 = blockIdx.x * BN;
  int bm0 = blockIdx.y * BM;
  int batch = blockIdx.z;
  float scale = 1.0f;
  if (MODE == 1 || MODE == 2) {
    int e = eidx[batch];
    Bmat += (long)e * strideBe;
    bias += (long)e * biasStrideE;
    A    += (long)batch * strideAb;
    if (MODE == 2) scale = gateW[batch];
  }

  int wave = tid >> 5, lane = tid & 31;
  int wm = wave & 1;    // 2 waves over M
  int wn = wave >> 1;   // 4 waves over N
  v8f acc[2][2] = {};

  unsigned asA = (unsigned)(uintptr_t)&As[0][0];
  unsigned asB = (unsigned)(uintptr_t)&Bs[0][0];
  const bf16* Atile = A + (long)bm0 * lda;   // + k
  const bf16* Btile = Bmat + bn0;            // + k*ldb

  int nsteps = K / BK;
  if (wave == 0) {   // prologue DMA: tile 0 into buffer 0 (TENSORcnt)
    tdm_load_2d(asA, Atile, BK, BM, (unsigned long)lda);
    tdm_load_2d(asB, Btile, BN, BK, (unsigned long)ldb);
  }
  for (int s = 0; s < nsteps; ++s) {
    int cur = s & 1, nxt = cur ^ 1;
    __syncthreads();  // everyone finished reading buf[nxt] (iter s-1)
    if (wave == 0) {
      if (s + 1 < nsteps) {
        tdm_load_2d(asA + nxt * ABYTES, Atile + (s + 1) * BK,
                    BK, BM, (unsigned long)lda);
        tdm_load_2d(asB + nxt * BBYTES, Btile + (long)(s + 1) * BK * ldb,
                    BN, BK, (unsigned long)ldb);
        __builtin_amdgcn_s_wait_tensorcnt(2);  // tile s complete, s+1 in flight
      } else {
        __builtin_amdgcn_s_wait_tensorcnt(0);
      }
    }
    __syncthreads();  // buf[cur] visible to all waves
    const bf16* Ab = &As[cur][0];
    const bf16* Bb = &Bs[cur][0];
    v16bf a0 = load_a16(&Ab[(wm * 32 + 0)  * BK], BK);
    v16bf a1 = load_a16(&Ab[(wm * 32 + 16) * BK], BK);
    v16bf b0 = load_b16(&Bb[wn * 32 + 0],  BN);
    v16bf b1 = load_b16(&Bb[wn * 32 + 16], BN);
    acc[0][0] = wmma_bf16(a0, b0, acc[0][0]);
    acc[0][1] = wmma_bf16(a0, b1, acc[0][1]);
    acc[1][0] = wmma_bf16(a1, b0, acc[1][0]);
    acc[1][1] = wmma_bf16(a1, b1, acc[1][1]);
  }

  // epilogue: C/D layout -> lane n = lane&15, rows m = r + 8*(lane>=16)
  int nlocal = lane & 15;
  int mhi    = (lane >> 4) << 3;
  #pragma unroll
  for (int i = 0; i < 2; ++i) {
    #pragma unroll
    for (int j = 0; j < 2; ++j) {
      int n = bn0 + wn * 32 + j * 16 + nlocal;
      float bv = bias ? bias[n] : 0.0f;
      #pragma unroll
      for (int r = 0; r < 8; ++r) {
        int m = bm0 + wm * 32 + i * 16 + mhi + r;
        float x = acc[i][j][r] + bv;
        long cidx = strideCb * batch + (long)m * ldc + n;
        if (MODE == 0) {
          Cf[cidx] = x;
        } else if (MODE == 3) {
          __builtin_nontemporal_store(x, &Cf[cidx]);  // 512MB stream: skip L2
        } else if (MODE == 1) {
          float g = 0.5f * x * (1.0f + erff(x * 0.70710678118654752f));
          Cbf[cidx] = (bf16)g;
        } else {
          Cbf[cidx] = (bf16)(x * scale);
        }
      }
    }
  }
}

// ---- RNN scan: 16 blocks x 128 threads, grid barrier per step -------------
__device__ __forceinline__ void grid_barrier(int* cnt, int* gen, int nblocks) {
  __syncthreads();
  if (threadIdx.x == 0) {
    __threadfence();
    int g = __atomic_load_n(gen, __ATOMIC_RELAXED);
    int arrived = __atomic_fetch_add(cnt, 1, __ATOMIC_ACQ_REL) + 1;
    if (arrived == nblocks) {
      __atomic_store_n(cnt, 0, __ATOMIC_RELAXED);
      __atomic_fetch_add(gen, 1, __ATOMIC_RELEASE);
    } else {
      while (__atomic_load_n(gen, __ATOMIC_ACQUIRE) == g)
        __builtin_amdgcn_s_sleep(1);
    }
    __threadfence();
  }
  __syncthreads();
}

#define SCAN_BLOCKS 16

__global__ __launch_bounds__(128) void rnn_scan_k(
    const float* __restrict__ U, const bf16* __restrict__ WhhT,
    bf16* __restrict__ hbuf /* 2 x 16*DIM, zeroed */,
    bf16* __restrict__ hsbf, float* __restrict__ meanH,
    int* barCnt, int* barGen) {
  int wave = threadIdx.x >> 5, lane = threadIdx.x & 31;
  int ntile = blockIdx.x * 4 + wave;  // 64 N-tiles cover D=1024
  int n0 = ntile * 16;
  int nlocal = lane & 15;
  int mhi = (lane >> 4) << 3;
  float meanAcc[8] = {};

  for (int t = 0; t < SEQ; ++t) {
    const bf16* hin  = hbuf + (size_t)(t & 1) * 16 * DIM;
    bf16*       hout = hbuf + (size_t)((t + 1) & 1) * 16 * DIM;
    v8f acc = {};
    for (int k0 = 0; k0 < DIM; k0 += 32) {
      v16bf a = load_a16(hin + k0, DIM);                    // h (16xD padded)
      v16bf b = load_b16(WhhT + (long)k0 * DIM + n0, DIM);  // loop-invariant
      acc = wmma_bf16(a, b, acc);
    }
    #pragma unroll
    for (int r = 0; r < 8; ++r) {
      int m = mhi + r;  // batch row
      if (m < NB) {
        int n = n0 + nlocal;
        float x = acc[r] + U[((long)m * SEQ + t) * DIM + n];
        float h = tanhf(x);
        meanAcc[r] += h;
        bf16 hb = (bf16)h;
        hout[m * DIM + n] = hb;
        hsbf[((long)m * SEQ + t) * DIM + n] = hb;
      }
    }
    grid_barrier(barCnt, barGen, SCAN_BLOCKS);
  }
  #pragma unroll
  for (int r = 0; r < 8; ++r) {
    int m = mhi + r;
    if (m < NB) meanH[m * DIM + n0 + nlocal] = meanAcc[r] * (1.0f / SEQ);
  }
}

// ---- gating: softmax(meanH @ Wg^T), top-1 ---------------------------------
__global__ __launch_bounds__(128) void gates_k(
    const float* __restrict__ meanH, const float* __restrict__ Wg,
    float* gateW, int* gateIdx) {
  int b = threadIdx.x >> 5, lane = threadIdx.x & 31;  // wave = batch
  float s[NE] = {};
  for (int d = lane; d < DIM; d += 32) {
    float x = meanH[b * DIM + d];
    #pragma unroll
    for (int e = 0; e < NE; ++e) s[e] += x * Wg[e * DIM + d];
  }
  #pragma unroll
  for (int e = 0; e < NE; ++e)
    #pragma unroll
    for (int off = 16; off > 0; off >>= 1)
      s[e] += __shfl_xor(s[e], off, 32);
  if (lane == 0) {
    float mx = s[0]; int bi = 0;
    #pragma unroll
    for (int e = 1; e < NE; ++e) if (s[e] > mx) { mx = s[e]; bi = e; }
    float denom = 0.0f;
    #pragma unroll
    for (int e = 0; e < NE; ++e) denom += expf(s[e] - mx);
    gateW[b] = 1.0f / denom;  // softmax value at argmax
    gateIdx[b] = bi;
  }
}

// ---------------------------------------------------------------------------
extern "C" void kernel_launch(void* const* d_in, const int* in_sizes, int n_in,
                              void* d_out, int out_size, void* d_ws, size_t ws_size,
                              hipStream_t stream) {
  const int*   tokens = (const int*)  d_in[0];
  const float* emb    = (const float*)d_in[1];
  const float* Wih    = (const float*)d_in[2];
  const float* bih    = (const float*)d_in[3];
  const float* Whh    = (const float*)d_in[4];
  const float* bhh    = (const float*)d_in[5];
  const float* Wg     = (const float*)d_in[6];
  const float* W1     = (const float*)d_in[7];
  const float* b1     = (const float*)d_in[8];
  const float* W2     = (const float*)d_in[9];
  const float* b2     = (const float*)d_in[10];
  const float* Wfc    = (const float*)d_in[11];
  const float* bfc    = (const float*)d_in[12];
  float* logits = (float*)d_out;

  char* ws = (char*)d_ws;
  auto alloc = [&](size_t bytes) {
    char* p = ws; ws += (bytes + 255) & ~(size_t)255; return p;
  };
  bf16*  Xbf   = (bf16*) alloc((size_t)NB * SEQ * DIM * 2);
  bf16*  hsbf  = (bf16*) alloc((size_t)NB * SEQ * DIM * 2);
  bf16*  Ybf   = (bf16*) alloc((size_t)NB * SEQ * DIM * 2);
  bf16*  H1bf  = (bf16*) alloc((size_t)NB * SEQ * DFF * 2);
  bf16*  WihT  = (bf16*) alloc((size_t)DIM * DIM * 2);
  bf16*  WhhT  = (bf16*) alloc((size_t)DIM * DIM * 2);
  bf16*  W1bf  = (bf16*) alloc((size_t)NE * DIM * DFF * 2);
  bf16*  W2bf  = (bf16*) alloc((size_t)NE * DFF * DIM * 2);
  bf16*  WfcT  = (bf16*) alloc((size_t)DIM * VOCAB * 2);
  float* U     = (float*)alloc((size_t)NB * SEQ * DIM * 4);
  bf16*  hbuf  = (bf16*) alloc((size_t)2 * 16 * DIM * 2);
  float* meanH = (float*)alloc((size_t)NB * DIM * 4);
  float* bias2 = (float*)alloc((size_t)DIM * 4);
  float* gateW = (float*)alloc(64);
  int*   gateIdx = (int*)alloc(64);
  int*   bars  = (int*)  alloc(64);  // [cnt, gen]

  // --- prep ---
  zero_u32_k<<<dim3((2 * 16 * DIM * 2 / 4 + 255) / 256), 256, 0, stream>>>(
      (unsigned int*)hbuf, 2 * 16 * DIM * 2 / 4);
  zero_u32_k<<<dim3(1), 256, 0, stream>>>((unsigned int*)bars, 2);
  bias2_k<<<dim3(DIM / 256), 256, 0, stream>>>(bih, bhh, bias2, DIM);
  embed_k<<<dim3(NB * SEQ, DIM / 256), 256, 0, stream>>>(tokens, emb, Xbf);

  {
    long n1 = (long)NE * DIM * DFF;
    cvt_bf_k<<<dim3((unsigned)(n1 / 4 / 256)), 256, 0, stream>>>(W1, W1bf, n1);
    long n2 = (long)NE * DFF * DIM;
    cvt_bf_k<<<dim3((unsigned)(n2 / 4 / 256)), 256, 0, stream>>>(W2, W2bf, n2);
  }
  transpose_bf_k<<<dim3(DIM / 32, DIM / 32), 256, 0, stream>>>(Wih, WihT, DIM, DIM);
  transpose_bf_k<<<dim3(DIM / 32, DIM / 32), 256, 0, stream>>>(Whh, WhhT, DIM, DIM);
  transpose_bf_k<<<dim3(DIM / 32, VOCAB / 32), 256, 0, stream>>>(Wfc, WfcT, VOCAB, DIM);

  // --- U = X @ Wih^T + (bih+bhh) : 4096x1024x1024, fp32 out ---
  gemm_bf16_k<0><<<dim3(DIM / BN, (NB * SEQ) / BM, 1), 256, 0, stream>>>(
      Xbf, WihT, bias2, U, nullptr,
      NB * SEQ, DIM, DIM, DIM, DIM, DIM,
      nullptr, 0, 0, 0, 0, nullptr);

  // --- sequential RNN scan across 16 WGPs ---
  rnn_scan_k<<<dim3(SCAN_BLOCKS), 128, 0, stream>>>(
      U, WhhT, hbuf, hsbf, meanH, &bars[0], &bars[1]);

  // --- gating / top-1 ---
  gates_k<<<dim3(1), 128, 0, stream>>>(meanH, Wg, gateW, gateIdx);

  // --- H1 = gelu(hs @ W1[e] + b1[e]) : per-batch 1024x1024x4096 ---
  gemm_bf16_k<1><<<dim3(DFF / BN, SEQ / BM, NB), 256, 0, stream>>>(
      hsbf, W1bf, b1, nullptr, H1bf,
      SEQ, DFF, DIM, DIM, DFF, DFF,
      gateIdx, (long)DIM * DFF, DFF, (long)SEQ * DIM, (long)SEQ * DFF, nullptr);

  // --- Y = (H1 @ W2[e] + b2[e]) * w : per-batch 1024x4096x1024 ---
  gemm_bf16_k<2><<<dim3(DIM / BN, SEQ / BM, NB), 256, 0, stream>>>(
      H1bf, W2bf, b2, nullptr, Ybf,
      SEQ, DIM, DFF, DFF, DIM, DIM,
      gateIdx, (long)DFF * DIM, DIM, (long)SEQ * DFF, (long)SEQ * DIM, gateW);

  // --- logits = Y @ Wfc^T + bfc : 4096x1024x32000, fp32 NT out (512 MB) ---
  gemm_bf16_k<3><<<dim3(VOCAB / BN, (NB * SEQ) / BM, 1), 256, 0, stream>>>(
      Ybf, WfcT, bfc, logits, nullptr,
      NB * SEQ, VOCAB, DIM, DIM, VOCAB, VOCAB,
      nullptr, 0, 0, 0, 0, nullptr);
}
